// ComposedCliffordSteerableKernel_45397804319202
// MI455X (gfx1250) — matrix operand order repaired and checked by reading, not compiled, not run on hardware
//
#include <hip/hip_runtime.h>
#include <hip/hip_bf16.h>

typedef _Float16 v16h __attribute__((ext_vector_type(16)));
typedef float    v8f  __attribute__((ext_vector_type(8)));

#define KS 15          // spatial kernel size
#define CH 256         // total channels = 64 * 4
#define SP 225         // 15*15
#define A_ROW 48       // bytes per k1pad row (32B data + 16B pad -> conflict-free)
#define A_BLK (29 * A_ROW)      // one (n,i) block: 29 padded rows
#define A_BYTES (16 * A_BLK)    // 22272
#define B_COL 1936     // bytes per Bexp column (60*32 + 16B pad -> conflict-free)
#define B_BYTES (64 * B_COL)    // 123904
#define SMEM_BYTES (A_BYTES + B_BYTES)  // 146176

__global__ __launch_bounds__(128)
void ccsk_wmma_kernel(const float* __restrict__ k1,
                      const float* __restrict__ k2,
                      float* __restrict__ out) {
  extern __shared__ char smem[];
  char* aBase = smem;            // k1pad: [ni=16][u=29][v=16] f16, y-pad of 7 rows each side
  char* bBase = smem + A_BYTES;  // Bexp:  [col=64][r=60][v=16] f16, Toeplitz x-shift expansion

  const int bx = blockIdx.x;
  const int co = bx >> 6;
  const int ci = bx & 63;
  const int tid = threadIdx.x;

  // ---------------- Phase 1: build k1pad (A operand, f16) ----------------
  // A[(n,y),(i,dy,v)] = k1[co4+n, ci4+i, y+dy-7, v]  -> pad rows u=y+dy in [0,29)
  for (int idx = tid; idx < 16 * 29; idx += 128) {
    const int ni = idx / 29, u = idx % 29;
    const int n = ni >> 2, i = ni & 3;
    _Float16 row[16];
#pragma unroll
    for (int v = 0; v < 16; ++v) row[v] = (_Float16)0.f;
    if (u >= 7 && u < 22) {
      const float* src = k1 + (size_t)((co * 4 + n) * CH + (ci * 4 + i)) * SP + (u - 7) * KS;
#pragma unroll
      for (int v = 0; v < 15; ++v) row[v] = (_Float16)src[v];
    }
    const uint32_t off = (uint32_t)(ni * A_BLK + u * A_ROW);
    *(uint4*)(aBase + off)      = *(const uint4*)(&row[0]);
    *(uint4*)(aBase + off + 16) = *(const uint4*)(&row[8]);
  }

  // ---------------- Phase 2: build Bexp (B operand, f16) ----------------
  // B[(i,dy,v),(o,x)] = k2[co4+o, ci4+i, dy, v-x+7] (0 outside) ; col = o*16+x
  for (int uu = tid; uu < 240; uu += 128) {
    const int o = uu / 60, r = uu % 60;        // r = i*15 + dy
    const int i = r / 15, dy = r % 15;
    _Float16 h[15];
    const float* src = k2 + (size_t)((co * 4 + o) * CH + (ci * 4 + i)) * SP + dy * KS;
#pragma unroll
    for (int v = 0; v < 15; ++v) h[v] = (_Float16)src[v];
#pragma unroll
    for (int x = 0; x < 16; ++x) {
      _Float16 w[16];
#pragma unroll
      for (int v = 0; v < 16; ++v) {
        const int j = v - x + 7;
        w[v] = (j >= 0 && j < 15) ? h[j] : (_Float16)0.f;
      }
      const uint32_t off = (uint32_t)((o * 16 + x) * B_COL + r * 32);
      *(uint4*)(bBase + off)      = *(const uint4*)(&w[0]);
      *(uint4*)(bBase + off + 16) = *(const uint4*)(&w[8]);
    }
  }

  __syncthreads();

  // ---------------- Phase 3: 64x64x960 GEMM via v_wmma_f32_16x16x32_f16 ----------------
  // 4 waves, each owns a 2x2 block of 16x16 tiles: 4 wmma per chunk off 8 ds_load_b128.
  const int lane = tid & 31;
  const int wid  = tid >> 5;
  const int tm0  = (wid >> 1) * 2;   // n-tiles tm0, tm0+1 (M)
  const int tn0  = (wid & 1) * 2;    // o-tiles tn0, tn0+1 (N)
  const int q    = lane & 15;        // y for A rows / x for B cols / col for C
  const int h    = lane >> 4;        // K-half selector per wave32 WMMA layout

  v8f acc00 = {0.f, 0.f, 0.f, 0.f, 0.f, 0.f, 0.f, 0.f};
  v8f acc01 = acc00, acc10 = acc00, acc11 = acc00;

  const char* bCol0 = bBase + (uint32_t)((tn0 * 16 + q) * B_COL + h * 16);
  const char* aLane = aBase + (uint32_t)(h * 16);

#pragma unroll
  for (int cc = 0; cc < 30; ++cc) {
    const int r0 = 2 * cc, r1 = r0 + 1;
    const int i0 = r0 / 15, dy0 = r0 % 15;   // constants after full unroll
    const int i1 = r1 / 15, dy1 = r1 % 15;

    // B fragments: elems 0..7 = row r0 @ v=h*8, elems 8..15 = row r1 @ v=h*8
    union { uint4 u[2]; v16h v; } bf0, bf1;
    const char* bp = bCol0 + (uint32_t)(r0 * 32);
    bf0.u[0] = *(const uint4*)(bp);
    bf0.u[1] = *(const uint4*)(bp + 32);
    bf1.u[0] = *(const uint4*)(bp + 16 * B_COL);
    bf1.u[1] = *(const uint4*)(bp + 16 * B_COL + 32);

    // A fragments: elems 0..7 = row r0 @ v=h*8, elems 8..15 = row r1 @ v=h*8
    union { uint4 u[2]; v16h v; } af0, af1;
    const uint32_t o0 = (uint32_t)(((tm0 * 4 + i0) * 29 + (q + dy0)) * A_ROW);
    const uint32_t o1 = (uint32_t)(((tm0 * 4 + i1) * 29 + (q + dy1)) * A_ROW);
    af0.u[0] = *(const uint4*)(aLane + o0);
    af0.u[1] = *(const uint4*)(aLane + o1);
    af1.u[0] = *(const uint4*)(aLane + o0 + 4 * A_BLK);
    af1.u[1] = *(const uint4*)(aLane + o1 + 4 * A_BLK);

    // 4 independent WMMAs back-to-back (no D->A/B RAW): hazard slots filled by issue
    acc00 = __builtin_amdgcn_wmma_f32_16x16x32_f16(false, af0.v, false, bf0.v,
                                                   (short)0, acc00, false, false);
    acc01 = __builtin_amdgcn_wmma_f32_16x16x32_f16(false, af0.v, false, bf1.v,
                                                   (short)0, acc01, false, false);
    acc10 = __builtin_amdgcn_wmma_f32_16x16x32_f16(false, af1.v, false, bf0.v,
                                                   (short)0, acc10, false, false);
    acc11 = __builtin_amdgcn_wmma_f32_16x16x32_f16(false, af1.v, false, bf1.v,
                                                   (short)0, acc11, false, false);
  }

  // ---------------- Store: C/D layout (lane = col, VGPR j = row j + 8h) ----------------
  if (q < 15) {
#pragma unroll
    for (int j = 0; j < 8; ++j) {
      const int m = j + h * 8;
      if (m < 15) {
        const size_t base = (size_t)m * KS + q;
        out[(size_t)((co * 4 + tm0)     * CH + (ci * 4 + tn0))     * SP + base] = acc00[j];
        out[(size_t)((co * 4 + tm0)     * CH + (ci * 4 + tn0 + 1)) * SP + base] = acc01[j];
        out[(size_t)((co * 4 + tm0 + 1) * CH + (ci * 4 + tn0))     * SP + base] = acc10[j];
        out[(size_t)((co * 4 + tm0 + 1) * CH + (ci * 4 + tn0 + 1)) * SP + base] = acc11[j];
      }
    }
  }
}

extern "C" void kernel_launch(void* const* d_in, const int* in_sizes, int n_in,
                              void* d_out, int out_size, void* d_ws, size_t ws_size,
                              hipStream_t stream) {
  const float* k1 = (const float*)d_in[0];
  const float* k2 = (const float*)d_in[1];
  float* out = (float*)d_out;

  static_assert(SMEM_BYTES == 146176, "LDS budget");
  hipFuncSetAttribute((const void*)ccsk_wmma_kernel,
                      hipFuncAttributeMaxDynamicSharedMemorySize, SMEM_BYTES);

  // one workgroup (4 wave32) per (co, ci) blade-block pair: 64*64 = 4096 blocks
  ccsk_wmma_kernel<<<dim3(4096), dim3(128), SMEM_BYTES, stream>>>(k1, k2, out);
}
